// RGCN_89781996355919
// MI455X (gfx1250) — matrix operand order repaired and checked by reading, not compiled, not run on hardware
//
#include <hip/hip_runtime.h>
#include <math.h>

typedef __attribute__((ext_vector_type(16))) _Float16 v16h;
typedef __attribute__((ext_vector_type(8)))  float    v8f;

#define RNUM 4
#define DK   128
#define TPB  256

// ---------------- utility kernels ----------------
__global__ void zero_f32(float* p, long n) {
  long i = (long)blockIdx.x * blockDim.x + threadIdx.x;
  if (i < n) p[i] = 0.0f;
}
__global__ void zero_i32(int* p, long n) {
  long i = (long)blockIdx.x * blockDim.x + threadIdx.x;
  if (i < n) p[i] = 0;
}
__global__ void tanh_k(float* p, long n) {
  long i = (long)blockIdx.x * blockDim.x + threadIdx.x;
  if (i < n) p[i] = tanhf(p[i]);
}

// ---------------- degree / inverse-degree ----------------
__global__ void count_k(const int* __restrict__ dst, const int* __restrict__ et,
                        int* __restrict__ cnt, int E) {
  int e = blockIdx.x * blockDim.x + threadIdx.x;
  if (e < E) atomicAdd(&cnt[dst[e] * RNUM + et[e]], 1);
}
__global__ void inv_k(const int* __restrict__ cnt, float* __restrict__ inv, int n) {
  int i = blockIdx.x * blockDim.x + threadIdx.x;
  if (i < n) { int c = cnt[i]; inv[i] = 1.0f / (float)(c > 0 ? c : 1); }
}

// ---------------- pack W (f32 [R,128,OUT]) into WMMA B-fragment order (f16) ----
// Fragment index: (((r*CT + c)*4 + kc)*32 + lane) * 16 halves, contiguous per lane.
// B order packed here (contiguous k, not AMD's table order): lane n<16 holds col n, K = kc*32 + 0..15;
// lanes 16-31 hold col n, K = kc*32 + 16..31.
__global__ void pack_w(const float* __restrict__ W, _Float16* __restrict__ WF,
                       int OUT, int CT) {
  int idx = blockIdx.x * blockDim.x + threadIdx.x;
  int total = RNUM * CT * 4 * 32;
  if (idx >= total) return;
  int lane = idx & 31;
  int kc   = (idx >> 5) & 3;
  int rest = idx >> 7;
  int c = rest % CT;
  int r = rest / CT;
  int n = lane & 15, hh = lane >> 4;
  v16h tmp;
  for (int i = 0; i < 16; ++i) {
    int k = kc * 32 + hh * 16 + i;
    int o = c * 16 + n;
    float v = (o < OUT) ? W[((size_t)r * DK + k) * OUT + o] : 0.0f;
    tmp[i] = (_Float16)v;
  }
  *(v16h*)(WF + (size_t)idx * 16) = tmp;
}

// ---------------- per-relation GEMM via WMMA, B staged in LDS ----------------
// Block = (relation = blockIdx.y, 8 row-tiles). B fragments for the relation
// (CT*4 fragments, 32B/lane) are async-copied global->LDS once per block
// (GLOBAL_LOAD_ASYNC_TO_LDS_B128, ASYNCcnt), then every wave feeds WMMA from
// ds_load_b128.
template <int CT, int OUT>
__global__ void gemm_wmma(const float* __restrict__ X, const _Float16* __restrict__ WF,
                          float* __restrict__ XW, int Nn) {
  constexpr int FRAG_HALVES = CT * 4 * 32 * 16;          // per relation
  __shared__ _Float16 BS[FRAG_HALVES];                   // 32 KB (CT=8) / 4 KB (CT=1)
  const int tid = threadIdx.x;
  const int r   = blockIdx.y;

  // ---- async stage B fragments into LDS ----
  {
    constexpr int chunks = FRAG_HALVES * 2 / 16;         // 16B chunks
    const char* gsrc = (const char*)(WF + (size_t)r * FRAG_HALVES);
    unsigned lbase = (unsigned)(uintptr_t)(&BS[0]);      // LDS byte offset (low 32 bits)
    for (int ch = tid; ch < chunks; ch += TPB) {
      unsigned laddr = lbase + (unsigned)ch * 16u;
      const char* ga = gsrc + (size_t)ch * 16;
      asm volatile("global_load_async_to_lds_b128 %0, %1, off"
                   :: "v"(laddr), "v"(ga) : "memory");
    }
    asm volatile("s_wait_asynccnt 0x0" ::: "memory");
    __syncthreads();
  }

  const int wave = tid >> 5, lane = tid & 31;
  const int ntiles = Nn >> 4;
  const int t = blockIdx.x * 8 + wave;
  if (t >= ntiles) return;                               // wave-uniform guard
  const int row0 = t << 4;
  const int m = lane & 15, hh = lane >> 4;

  // A fragment (16-bit 16x32): lanes 0-15 = rows M, lanes 16-31 same rows,
  // VGPR vg -> K = vg*2 + hh*16 (contiguous, matching packed B).
  const float* xr = X + (size_t)(row0 + m) * DK;
  v16h afrag[4];
  for (int kc = 0; kc < 4; ++kc)
    for (int vg = 0; vg < 8; ++vg) {
      int kk = kc * 32 + hh * 16 + vg * 2;
      afrag[kc][2 * vg]     = (_Float16)xr[kk];
      afrag[kc][2 * vg + 1] = (_Float16)xr[kk + 1];
    }

  const int nn = lane & 15;
  for (int c = 0; c < CT; ++c) {
    v8f acc = {};
    for (int kc = 0; kc < 4; ++kc) {
      v16h b = *(const v16h*)(&BS[(size_t)(((c * 4 + kc) * 32) + lane) * 16]);
      acc = __builtin_amdgcn_wmma_f32_16x16x32_f16(false, afrag[kc], false, b,
                                                   (short)0, acc, false, false);
    }
    // C/D layout: VGPR j, lanes 0-15 -> M=j, lanes 16-31 -> M=j+8, N=lane&15
    for (int j = 0; j < 8; ++j) {
      const int col  = c * 16 + nn;
      const int mrow = row0 + j + hh * 8;
      if constexpr (OUT >= CT * 16) {
        XW[((size_t)r * Nn + mrow) * OUT + col] = acc[j];  // guard compile-time dead
      } else {
        if (col < OUT) XW[((size_t)r * Nn + mrow) * OUT + col] = acc[j];
      }
    }
  }
}

// ---------------- edge scatter, d_out = 128: one wave per edge -------------
__global__ void scatter_d128(const float* __restrict__ XW,
                             const int* __restrict__ src, const int* __restrict__ dst,
                             const int* __restrict__ et, const float* __restrict__ inv,
                             float* __restrict__ out, int E, int Nn) {
  int e    = (int)(((long)blockIdx.x * blockDim.x + threadIdx.x) >> 5);
  int lane = threadIdx.x & 31;
  if (e >= E) return;
  int s = src[e], d = dst[e], r = et[e];
  float sc = inv[d * RNUM + r];
  const float4* p = (const float4*)(XW + ((size_t)r * Nn + s) * DK);
  float4 v = p[lane];
  float* o = out + (size_t)d * DK + lane * 4;
  atomicAdd(o + 0, v.x * sc);
  atomicAdd(o + 1, v.y * sc);
  atomicAdd(o + 2, v.z * sc);
  atomicAdd(o + 3, v.w * sc);
}

// ---------------- edge scatter, d_out = 2: one thread per edge -------------
__global__ void scatter_d2(const float* __restrict__ XW,
                           const int* __restrict__ src, const int* __restrict__ dst,
                           const int* __restrict__ et, const float* __restrict__ inv,
                           float* __restrict__ out, int E, int Nn) {
  int e = blockIdx.x * blockDim.x + threadIdx.x;
  if (e >= E) return;
  int s = src[e], d = dst[e], r = et[e];
  float sc = inv[d * RNUM + r];
  const float* p = XW + ((size_t)r * Nn + s) * 2;
  atomicAdd(out + (size_t)d * 2 + 0, p[0] * sc);
  atomicAdd(out + (size_t)d * 2 + 1, p[1] * sc);
}

// ---------------- softmax over C=2 + logits ----------------
__global__ void softmax_k(const float* __restrict__ h, float* __restrict__ out, int n) {
  int i = blockIdx.x * blockDim.x + threadIdx.x;
  if (i >= n) return;
  float a = h[2 * i], b = h[2 * i + 1];
  float mx = fmaxf(a, b);
  float ea = __expf(a - mx), eb = __expf(b - mx);
  float s = ea + eb;
  out[2 * i]     = ea / s;
  out[2 * i + 1] = eb / s;
  out[(size_t)2 * n + 2 * i]     = a;
  out[(size_t)2 * n + 2 * i + 1] = b;
}

extern "C" void kernel_launch(void* const* d_in, const int* in_sizes, int n_in,
                              void* d_out, int out_size, void* d_ws, size_t ws_size,
                              hipStream_t stream) {
  const float* x  = (const float*)d_in[0];
  const int*   ei = (const int*)d_in[1];
  const int*   et = (const int*)d_in[2];
  const float* W1 = (const float*)d_in[3];
  const float* W2 = (const float*)d_in[4];
  const float* W3 = (const float*)d_in[5];
  float* out = (float*)d_out;

  const int N = in_sizes[0] / DK;   // 100000
  const int E = in_sizes[2];        // 1600000
  const int* src  = ei;
  const int* dstp = ei + E;

  // workspace carve-up
  char* ws = (char*)d_ws;
  size_t off = 0;
  auto alloc = [&](size_t bytes) -> void* {
    void* p = ws + off;
    off = (off + bytes + 255) & ~(size_t)255;
    return p;
  };
  float*    XW   = (float*)alloc((size_t)RNUM * N * DK * sizeof(float));   // 204.8 MB
  float*    hA   = (float*)alloc((size_t)N * DK * sizeof(float));          // 51.2 MB
  float*    hB   = (float*)alloc((size_t)N * DK * sizeof(float));          // 51.2 MB
  _Float16* WF   = (_Float16*)alloc((size_t)RNUM * 8 * 4 * 32 * 16 * sizeof(_Float16));
  int*      cnt  = (int*)alloc((size_t)N * RNUM * sizeof(int));
  float*    inv  = (float*)alloc((size_t)N * RNUM * sizeof(float));
  float*    agg3 = (float*)alloc((size_t)N * 2 * sizeof(float));
  (void)ws_size; (void)n_in; (void)out_size;

  const long nr = (long)N * RNUM;
  const long nd = (long)N * DK;
  const int  ntiles = N / 16;                                 // 6250
  const dim3 gemmGrid((ntiles + 7) / 8, RNUM);
  const int  pack8 = RNUM * 8 * 4 * 32, pack1 = RNUM * 1 * 4 * 32;
  const int  scatBlocks = (E + 7) / 8;                        // 8 edges (waves) / block

  // degrees (graph is layer-invariant)
  zero_i32<<<(int)((nr + TPB - 1) / TPB), TPB, 0, stream>>>(cnt, nr);
  count_k<<<(E + TPB - 1) / TPB, TPB, 0, stream>>>(dstp, et, cnt, E);
  inv_k<<<(int)((nr + TPB - 1) / TPB), TPB, 0, stream>>>(cnt, inv, (int)nr);

  // ---- layer 1 ----
  pack_w<<<(pack8 + TPB - 1) / TPB, TPB, 0, stream>>>(W1, WF, DK, 8);
  gemm_wmma<8, DK><<<gemmGrid, TPB, 0, stream>>>(x, WF, XW, N);
  zero_f32<<<(int)((nd + TPB - 1) / TPB), TPB, 0, stream>>>(hA, nd);
  scatter_d128<<<scatBlocks, TPB, 0, stream>>>(XW, src, dstp, et, inv, hA, E, N);
  tanh_k<<<(int)((nd + TPB - 1) / TPB), TPB, 0, stream>>>(hA, nd);

  // ---- layer 2 ----
  pack_w<<<(pack8 + TPB - 1) / TPB, TPB, 0, stream>>>(W2, WF, DK, 8);
  gemm_wmma<8, DK><<<gemmGrid, TPB, 0, stream>>>(hA, WF, XW, N);
  zero_f32<<<(int)((nd + TPB - 1) / TPB), TPB, 0, stream>>>(hB, nd);
  scatter_d128<<<scatBlocks, TPB, 0, stream>>>(XW, src, dstp, et, inv, hB, E, N);
  tanh_k<<<(int)((nd + TPB - 1) / TPB), TPB, 0, stream>>>(hB, nd);

  // ---- layer 3 (d_out = 2, padded to one 16-wide WMMA tile) ----
  pack_w<<<(pack1 + TPB - 1) / TPB, TPB, 0, stream>>>(W3, WF, 2, 1);
  gemm_wmma<1, 2><<<gemmGrid, TPB, 0, stream>>>(hB, WF, XW, N);
  zero_f32<<<(int)(((long)N * 2 + TPB - 1) / TPB), TPB, 0, stream>>>(agg3, (long)N * 2);
  scatter_d2<<<(E + TPB - 1) / TPB, TPB, 0, stream>>>(XW, src, dstp, et, inv, agg3, E, N);
  softmax_k<<<(N + TPB - 1) / TPB, TPB, 0, stream>>>(agg3, out, N);
}
